// EQUICAT_37469294691144
// MI455X (gfx1250) — matrix operand order repaired and checked by reading, not compiled, not run on hardware
//
#include <hip/hip_runtime.h>
#include <hip/hip_bf16.h>

// ---------------------------------------------------------------------------
// MACE-style equivariant GNN layer stack for MI455X (gfx1250, wave32, WMMA).
// ---------------------------------------------------------------------------

#define DEV __device__ __forceinline__

constexpr int NN   = 50000;    // nodes
constexpr int EE   = 800000;   // edges
constexpr int C    = 32;       // channels
constexpr int NEL  = 4;        // elements
constexpr int LIT  = 2;        // interactions
constexpr int NB   = 8;        // bessel fns
constexpr float RCUT      = 5.0f;
constexpr float PI_F      = 3.14159265358979323846f;
constexpr float SQRT3     = 1.7320508075688772f;
constexpr float INV_SQRT3 = 1.0f / SQRT3;
constexpr float INV_SQRT2 = 0.7071067811865476f;

typedef __attribute__((ext_vector_type(2)))  float  v2f;
typedef __attribute__((ext_vector_type(8)))  float  v8f;
typedef __attribute__((ext_vector_type(16))) __bf16 v16bf;

DEV v8f wmma_f32x4(v2f a, v2f b, v8f c) {
  // V_WMMA_F32_16X16X4_F32
  return __builtin_amdgcn_wmma_f32_16x16x4_f32(false, a, false, b, (short)0, c,
                                               false, false);
}
DEV v8f wmma_bf16x32(v16bf a, v16bf b, v8f c) {
  // V_WMMA_F32_16X16X32_BF16
  return __builtin_amdgcn_wmma_f32_16x16x32_bf16(false, a, false, b, (short)0,
                                                 c, false, false);
}

DEV float silu(float x) { return x / (1.0f + __expf(-x)); }
DEV v8f v8f_zero() {
  v8f r;
#pragma unroll
  for (int j = 0; j < 8; ++j) r[j] = 0.0f;
  return r;
}

// ---- f32 WMMA GEMM tile: D[16, NT*16] += A[16, K4*4] * B[K4*4, NT*16] ------
// A layout (16x4 f32): lanes 0-15 hold M=0..15; lane-half selects K pair.
// B layout (4x16 f32): lane-half selects K pair; lo = column.
template <int K4, int NT, class FA, class FB>
DEV void gemm_f32(FA loadA, FB loadB, v8f* acc) {
  const int lo = threadIdx.x & 15;
  const int hi = (threadIdx.x >> 4) & 1;
#pragma unroll
  for (int kk = 0; kk < K4; ++kk) {
    const int k0 = kk * 4 + 2 * hi;
    v2f a;
    a.x = loadA(lo, k0);
    a.y = loadA(lo, k0 + 1);
#pragma unroll
    for (int nt = 0; nt < NT; ++nt) {
      v2f b;
      b.x = loadB(k0, nt * 16 + lo);
      b.y = loadB(k0 + 1, nt * 16 + lo);
      acc[nt] = wmma_f32x4(a, b, acc[nt]);
    }
  }
}

// ---- bf16 WMMA GEMM tile: D[16, NT*16] += A[16, KB*32] * B[KB*32, NT*16] ---
// 16-bit A 16x32 layout: vgpr v<4 -> K = 2v+j (+8 for lane-half 1);
//                        vgpr v>=4 -> K = 16 + 2(v-4)+j (+8 for lane-half 1).
// 16-bit B 32x16 layout: same K mapping as A (elem t<8 -> K=t+8h, t>=8 -> K=16+(t-8)+8h); n=lo.
template <int KB, int NT, class FA, class FB>
DEV void gemm_bf16(FA loadA, FB loadB, v8f* acc) {
  const int lo = threadIdx.x & 15;
  const int hi = (threadIdx.x >> 4) & 1;
#pragma unroll
  for (int kb = 0; kb < KB; ++kb) {
    v16bf a;
#pragma unroll
    for (int t = 0; t < 16; ++t) {
      const int v = t >> 1, j = t & 1;
      const int k = (v < 4) ? (2 * v + j + 8 * hi)
                            : (16 + 2 * (v - 4) + j + 8 * hi);
      a[t] = loadA(lo, kb * 32 + k);
    }
#pragma unroll
    for (int nt = 0; nt < NT; ++nt) {
      v16bf b;
#pragma unroll
      for (int t = 0; t < 16; ++t)
        b[t] = loadB(kb * 32 + ((t < 8) ? (8 * hi + t) : (16 + 8 * hi + t - 8)), nt * 16 + lo);
      acc[nt] = wmma_bf16x32(a, b, acc[nt]);
    }
  }
}

// ---------------------------------------------------------------------------
// Small utility kernels
// ---------------------------------------------------------------------------
__global__ void zero_kernel(float* __restrict__ p, int n) {
  const int i = blockIdx.x * blockDim.x + threadIdx.x;
  if (i < n) p[i] = 0.0f;
}

__global__ void cvt_bf16_kernel(const float* __restrict__ src,
                                __bf16* __restrict__ dst, int n) {
  const int i = blockIdx.x * blockDim.x + threadIdx.x;
  if (i < n) dst[i] = (__bf16)src[i];
}

__global__ void init_nodes_kernel(const int* __restrict__ species,
                                  const float* __restrict__ w_embed,
                                  float* __restrict__ f0,
                                  float* __restrict__ f1) {
  const int idx = blockIdx.x * blockDim.x + threadIdx.x;
  if (idx >= NN * C) return;
  const int n = idx / C, c = idx % C;
  f0[idx] = w_embed[species[n] * C + c];
  f1[(size_t)idx * 3 + 0] = 0.0f;
  f1[(size_t)idx * 3 + 1] = 0.0f;
  f1[(size_t)idx * 3 + 2] = 0.0f;
}

__global__ void writeout_kernel(const float* __restrict__ f0,
                                const float* __restrict__ f1,
                                float* __restrict__ out) {
  const int idx = blockIdx.x * blockDim.x + threadIdx.x;
  if (idx >= NN * C) return;
  out[(size_t)idx * 4 + 0] = f0[idx];
  out[(size_t)idx * 4 + 1] = f1[(size_t)idx * 3 + 0];
  out[(size_t)idx * 4 + 2] = f1[(size_t)idx * 3 + 1];
  out[(size_t)idx * 4 + 3] = f1[(size_t)idx * 3 + 2];
}

// ---------------------------------------------------------------------------
// Edge geometry: unit vectors * sqrt3 (l=1 SH) + Bessel * polynomial cutoff
// ---------------------------------------------------------------------------
__global__ void edge_geom_kernel(const float* __restrict__ pos,
                                 const int* __restrict__ EI,
                                 float* __restrict__ y1,
                                 float* __restrict__ ef) {
  const int e = blockIdx.x * blockDim.x + threadIdx.x;
  if (e >= EE) return;
  const int s = EI[e], r = EI[EE + e];
  const float dx = pos[(size_t)r * 3 + 0] - pos[(size_t)s * 3 + 0];
  const float dy = pos[(size_t)r * 3 + 1] - pos[(size_t)s * 3 + 1];
  const float dz = pos[(size_t)r * 3 + 2] - pos[(size_t)s * 3 + 2];
  const float len = sqrtf(dx * dx + dy * dy + dz * dz);
  const float inv = 1.0f / (len + 1e-9f);
  y1[(size_t)e * 3 + 0] = SQRT3 * dx * inv;
  y1[(size_t)e * 3 + 1] = SQRT3 * dy * inv;
  y1[(size_t)e * 3 + 2] = SQRT3 * dz * inv;
  const float u = len / RCUT;
  float u2 = u * u;
  float u6 = u2 * u2 * u2;
  float env = (u < 1.0f)
                  ? (1.0f - 28.0f * u6 + 48.0f * u6 * u - 21.0f * u6 * u2)
                  : 0.0f;
  const float base = 0.6324555320336759f /* sqrt(2/RCUT) */ * inv * env;
  const float arg = PI_F * len / RCUT;
#pragma unroll
  for (int k = 1; k <= NB; ++k)
    ef[(size_t)e * NB + (k - 1)] = base * __sinf((float)k * arg);
}

// ---------------------------------------------------------------------------
// linear_up: g0 = f0 @ W[0];  g1[:,:,x] = f1[:,:,x] @ W[1]   (f32 WMMA)
// grid: (N/16, 4 planes), block: 32 (one wave)
// ---------------------------------------------------------------------------
__global__ void node_up_kernel(const float* __restrict__ f0,
                               const float* __restrict__ f1,
                               const float* __restrict__ Wup,  // [2][32][32]
                               float* __restrict__ g0,
                               float* __restrict__ g1) {
  const int row0 = blockIdx.x * 16;
  const int plane = blockIdx.y;  // 0: scalar irrep, 1..3: vector components
  const float* W = Wup + (plane ? 1 : 0) * C * C;
  v8f acc[2] = {v8f_zero(), v8f_zero()};
  auto lb = [&](int k, int n) { return W[k * C + n]; };
  if (plane == 0) {
    auto la = [&](int m, int k) { return f0[(size_t)(row0 + m) * C + k]; };
    gemm_f32<8, 2>(la, lb, acc);
  } else {
    const int x = plane - 1;
    auto la = [&](int m, int k) {
      return f1[((size_t)(row0 + m) * C + k) * 3 + x];
    };
    gemm_f32<8, 2>(la, lb, acc);
  }
  const int lo = threadIdx.x & 15, hi = (threadIdx.x >> 4) & 1;
#pragma unroll
  for (int nt = 0; nt < 2; ++nt)
#pragma unroll
    for (int v = 0; v < 8; ++v) {
      const int r = row0 + v + 8 * hi;
      const int c = nt * 16 + lo;
      if (plane == 0)
        g0[(size_t)r * C + c] = acc[nt][v];
      else
        g1[((size_t)r * C + c) * 3 + (plane - 1)] = acc[nt][v];
    }
}

// ---------------------------------------------------------------------------
// Fused edge kernel: 3-layer radial MLP (WMMA) -> CG tensor product messages
// -> atomic scatter into receivers.  One wave per 16-edge tile.
// ---------------------------------------------------------------------------
__global__ void edge_msg_kernel(const int* __restrict__ EI,
                                const float* __restrict__ y1,
                                const float* __restrict__ ef,
                                const float* __restrict__ W1,   // f32 [8][64]
                                const __bf16* __restrict__ W2b, // bf16 [64][64]
                                const __bf16* __restrict__ W3b, // bf16 [64][160]
                                const float* __restrict__ g0,
                                const float* __restrict__ g1,
                                float* __restrict__ M0,
                                float* __restrict__ M1) {
  __shared__ __bf16 h1[16 * 64];
  __shared__ __bf16 h2[16 * 64];
  __shared__ float  wbuf[16 * 160];

  const int e0 = blockIdx.x * 16;
  const int lo = threadIdx.x & 15, hi = (threadIdx.x >> 4) & 1;

  // ---- layer 1: [16,8] @ [8,64], f32 WMMA (K=8) ----
  {
    v8f acc[4] = {v8f_zero(), v8f_zero(), v8f_zero(), v8f_zero()};
    auto la = [&](int m, int k) {
      const int e = e0 + m;
      return (e < EE) ? ef[(size_t)e * NB + k] : 0.0f;
    };
    auto lb = [&](int k, int n) { return W1[k * 64 + n]; };
    gemm_f32<2, 4>(la, lb, acc);
#pragma unroll
    for (int nt = 0; nt < 4; ++nt)
#pragma unroll
      for (int v = 0; v < 8; ++v)
        h1[(v + 8 * hi) * 64 + nt * 16 + lo] = (__bf16)silu(acc[nt][v]);
  }
  __syncthreads();

  // ---- layer 2: [16,64] @ [64,64], bf16 WMMA ----
  {
    v8f acc[4] = {v8f_zero(), v8f_zero(), v8f_zero(), v8f_zero()};
    auto la = [&](int m, int k) { return h1[m * 64 + k]; };
    auto lb = [&](int k, int n) { return W2b[k * 64 + n]; };
    gemm_bf16<2, 4>(la, lb, acc);
#pragma unroll
    for (int nt = 0; nt < 4; ++nt)
#pragma unroll
      for (int v = 0; v < 8; ++v)
        h2[(v + 8 * hi) * 64 + nt * 16 + lo] = (__bf16)silu(acc[nt][v]);
  }
  __syncthreads();

  // ---- layer 3: [16,64] @ [64,160], bf16 WMMA ----
  {
    v8f acc[10];
#pragma unroll
    for (int nt = 0; nt < 10; ++nt) acc[nt] = v8f_zero();
    auto la = [&](int m, int k) { return h2[m * 64 + k]; };
    auto lb = [&](int k, int n) { return W3b[k * 160 + n]; };
    gemm_bf16<2, 10>(la, lb, acc);
#pragma unroll
    for (int nt = 0; nt < 10; ++nt)
#pragma unroll
      for (int v = 0; v < 8; ++v)
        wbuf[(v + 8 * hi) * 160 + nt * 16 + lo] = acc[nt][v];
  }
  __syncthreads();

  // ---- message phase: lane = channel, iterate over the 16 edges ----
  const int c = threadIdx.x;  // 0..31
  for (int le = 0; le < 16; ++le) {
    const int e = e0 + le;
    if (e >= EE) break;
    const int s = EI[e];
    const int r = EI[EE + e];
    // prefetch next sender's feature rows while we work on this edge
    if (le + 1 < 16 && e + 1 < EE) {
      const int sn = EI[e + 1];
      __builtin_prefetch(&g0[(size_t)sn * C], 0, 0);
      __builtin_prefetch(&g1[(size_t)sn * C * 3], 0, 0);
    }
    const float yx = y1[(size_t)e * 3 + 0];
    const float yy = y1[(size_t)e * 3 + 1];
    const float yz = y1[(size_t)e * 3 + 2];
    const float x0v = g0[(size_t)s * C + c];
    const size_t bs = ((size_t)s * C + c) * 3;
    const float x1x = g1[bs + 0], x1y = g1[bs + 1], x1z = g1[bs + 2];
    const float w0 = wbuf[le * 160 + 0 * C + c];
    const float w1 = wbuf[le * 160 + 1 * C + c];
    const float w2 = wbuf[le * 160 + 2 * C + c];
    const float w3 = wbuf[le * 160 + 3 * C + c];
    const float w4 = wbuf[le * 160 + 4 * C + c];
    const float dot = x1x * yx + x1y * yy + x1z * yz;
    const float m0 = w0 * x0v + w3 * dot * INV_SQRT3;
    const float cx = x1y * yz - x1z * yy;
    const float cy = x1z * yx - x1x * yz;
    const float cz = x1x * yy - x1y * yx;
    const float m1x = w1 * x0v * yx + w2 * x1x + w4 * cx * INV_SQRT2;
    const float m1y = w1 * x0v * yy + w2 * x1y + w4 * cy * INV_SQRT2;
    const float m1z = w1 * x0v * yz + w2 * x1z + w4 * cz * INV_SQRT2;
    atomicAdd(&M0[(size_t)r * C + c], m0);
    const size_t br = ((size_t)r * C + c) * 3;
    atomicAdd(&M1[br + 0], m1x);
    atomicAdd(&M1[br + 1], m1y);
    atomicAdd(&M1[br + 2], m1z);
  }
}

// ---------------------------------------------------------------------------
// Fused node post: interaction linear -> element product basis -> out linear
// One wave per 16-node tile; all 4 planes handled by the wave.
// ---------------------------------------------------------------------------
__global__ void node_post_kernel(const float* __restrict__ M0,
                                 const float* __restrict__ M1,
                                 const float* __restrict__ Wlin,  // [2][32][32]
                                 const float* __restrict__ Wprod, // [NEL][32][5]
                                 const float* __restrict__ Wout,  // [2][32][32]
                                 const int* __restrict__ species,
                                 float* __restrict__ f0,
                                 float* __restrict__ f1) {
  __shared__ float obuf[4][16 * C];
  const int row0 = blockIdx.x * 16;
  const int lo = threadIdx.x & 15, hi = (threadIdx.x >> 4) & 1;

  // ---- interaction linear (f32 WMMA) ----
  v8f a0[2] = {v8f_zero(), v8f_zero()};
  v8f a1[3][2];
#pragma unroll
  for (int x = 0; x < 3; ++x) {
    a1[x][0] = v8f_zero();
    a1[x][1] = v8f_zero();
  }
  {
    auto la = [&](int m, int k) { return M0[(size_t)(row0 + m) * C + k]; };
    auto lb = [&](int k, int n) { return Wlin[k * C + n]; };
    gemm_f32<8, 2>(la, lb, a0);
  }
#pragma unroll
  for (int x = 0; x < 3; ++x) {
    auto la = [&](int m, int k) {
      return M1[((size_t)(row0 + m) * C + k) * 3 + x];
    };
    auto lb = [&](int k, int n) { return Wlin[C * C + k * C + n]; };
    gemm_f32<8, 2>(la, lb, a1[x]);
  }

  // ---- element-dependent symmetric product basis (correlation 2) ----
#pragma unroll
  for (int nt = 0; nt < 2; ++nt)
#pragma unroll
    for (int v = 0; v < 8; ++v) {
      const int rloc = v + 8 * hi;
      const int r = row0 + rloc;
      const int c = nt * 16 + lo;
      const int sp = species[r];
      const float* wp = Wprod + ((size_t)sp * C + c) * 5;
      const float t0 = a0[nt][v];
      const float tx = a1[0][nt][v], ty = a1[1][nt][v], tz = a1[2][nt][v];
      obuf[0][rloc * C + c] =
          wp[0] * t0 + wp[1] * t0 * t0 + wp[2] * (tx * tx + ty * ty + tz * tz);
      obuf[1][rloc * C + c] = wp[3] * tx + wp[4] * t0 * tx;
      obuf[2][rloc * C + c] = wp[3] * ty + wp[4] * t0 * ty;
      obuf[3][rloc * C + c] = wp[3] * tz + wp[4] * t0 * tz;
    }
  __syncthreads();

  // ---- product-layer linear (f32 WMMA) ----
  {
    v8f d[2] = {v8f_zero(), v8f_zero()};
    auto la = [&](int m, int k) { return obuf[0][m * C + k]; };
    auto lb = [&](int k, int n) { return Wout[k * C + n]; };
    gemm_f32<8, 2>(la, lb, d);
#pragma unroll
    for (int nt = 0; nt < 2; ++nt)
#pragma unroll
      for (int v = 0; v < 8; ++v)
        f0[(size_t)(row0 + v + 8 * hi) * C + nt * 16 + lo] = d[nt][v];
  }
#pragma unroll
  for (int x = 0; x < 3; ++x) {
    v8f d[2] = {v8f_zero(), v8f_zero()};
    auto la = [&](int m, int k) { return obuf[1 + x][m * C + k]; };
    auto lb = [&](int k, int n) { return Wout[C * C + k * C + n]; };
    gemm_f32<8, 2>(la, lb, d);
#pragma unroll
    for (int nt = 0; nt < 2; ++nt)
#pragma unroll
      for (int v = 0; v < 8; ++v)
        f1[((size_t)(row0 + v + 8 * hi) * C + nt * 16 + lo) * 3 + x] =
            d[nt][v];
  }
}

// ---------------------------------------------------------------------------
// Host launch
// ---------------------------------------------------------------------------
extern "C" void kernel_launch(void* const* d_in, const int* in_sizes, int n_in,
                              void* d_out, int out_size, void* d_ws,
                              size_t ws_size, hipStream_t stream) {
  const float* positions = (const float*)d_in[0];
  const int*   species   = (const int*)d_in[1];
  const int*   edge_idx  = (const int*)d_in[2];
  const float* w_embed   = (const float*)d_in[3];
  const float* w_up      = (const float*)d_in[4];
  const float* mlp_w1    = (const float*)d_in[5];
  const float* mlp_w2    = (const float*)d_in[6];
  const float* mlp_w3    = (const float*)d_in[7];
  const float* w_lin     = (const float*)d_in[8];
  const float* w_prod    = (const float*)d_in[9];
  const float* w_out     = (const float*)d_in[10];
  float* out = (float*)d_out;

  const size_t NC = (size_t)NN * C;
  float* fp = (float*)d_ws;
  float* f0  = fp; fp += NC;
  float* f1  = fp; fp += NC * 3;
  float* g0  = fp; fp += NC;
  float* g1  = fp; fp += NC * 3;
  float* M0  = fp; fp += NC;       // M0 and M1 contiguous (single zero pass)
  float* M1  = fp; fp += NC * 3;
  float* y1b = fp; fp += (size_t)EE * 3;
  float* efb = fp; fp += (size_t)EE * NB;
  __bf16* w2b = (__bf16*)fp;
  __bf16* w3b = w2b + (size_t)LIT * 64 * 64;

  // one-time conversions + embeddings + edge geometry
  cvt_bf16_kernel<<<(LIT * 64 * 64 + 255) / 256, 256, 0, stream>>>(
      mlp_w2, w2b, LIT * 64 * 64);
  cvt_bf16_kernel<<<(LIT * 64 * 160 + 255) / 256, 256, 0, stream>>>(
      mlp_w3, w3b, LIT * 64 * 160);
  init_nodes_kernel<<<(NN * C + 255) / 256, 256, 0, stream>>>(species, w_embed,
                                                              f0, f1);
  edge_geom_kernel<<<(EE + 255) / 256, 256, 0, stream>>>(positions, edge_idx,
                                                         y1b, efb);

  for (int i = 0; i < LIT; ++i) {
    node_up_kernel<<<dim3(NN / 16, 4), 32, 0, stream>>>(
        f0, f1, w_up + (size_t)i * 2 * C * C, g0, g1);
    zero_kernel<<<((int)(NC * 4) + 255) / 256, 256, 0, stream>>>(M0,
                                                                 (int)(NC * 4));
    edge_msg_kernel<<<EE / 16, 32, 0, stream>>>(
        edge_idx, y1b, efb, mlp_w1 + (size_t)i * NB * 64,
        w2b + (size_t)i * 64 * 64, w3b + (size_t)i * 64 * 160, g0, g1, M0, M1);
    node_post_kernel<<<NN / 16, 32, 0, stream>>>(
        M0, M1, w_lin + (size_t)i * 2 * C * C,
        w_prod + (size_t)i * NEL * C * 5, w_out + (size_t)i * 2 * C * C,
        species, f0, f1);
  }
  writeout_kernel<<<(NN * C + 255) / 256, 256, 0, stream>>>(f0, f1, out);
}
